// TriPathMultiScaleCrossAttention_40939628266031
// MI455X (gfx1250) — compile-verified, matrix operand order checked by reading, not run on hardware
//
#include <hip/hip_runtime.h>

typedef __attribute__((ext_vector_type(16))) __bf16 v16bf;
typedef __attribute__((ext_vector_type(8)))  float  v8f;
typedef __attribute__((ext_vector_type(2)))  float  v2f;
typedef __attribute__((ext_vector_type(4)))  int    v4i;

#define DEVI __device__ __forceinline__

constexpr int BATCH = 16;
constexpr int CD    = 256;
constexpr int HW    = 9216;
constexpr float INV_SCALE = 0.17677669529663687f;  // 1/sqrt(32)
constexpr float EPS = 1e-5f;

DEVI int ECP(int p)  { return p == 0 ? 128 : (p == 1 ? 256 : 512); }
DEVI int PREF(int p) { return p == 0 ? 0 : (p == 1 ? 128 : 384); }
DEVI size_t GOFF(int p, int b) {
  return (size_t)BATCH * CD * PREF(p) + (size_t)b * CD * ECP(p);
}
DEVI int SFO(int p, int b) { return BATCH * PREF(p) + b * ECP(p); }

// 224 tiles per batch: p0 -> 8x4, p1 -> 8x8, p2 -> 8x16
DEVI void decode224(int t, int& p, int& a, int& ni) {
  if (t < 32)      { p = 0; }
  else if (t < 96) { p = 1; t -= 32; }
  else             { p = 2; t -= 96; }
  int et = ECP(p) >> 5;
  a = t / et; ni = t % et;
}

DEVI v16bf cvt16(const float* f) {
  v16bf r;
#pragma unroll
  for (int i = 0; i < 16; ++i) r[i] = (__bf16)f[i];
  return r;
}

DEVI v16bf ld_bf16x16(const __bf16* p) {
  union { uint4 u[2]; v16bf v; } x;
  x.u[0] = *(const uint4*)p;
  x.u[1] = *(const uint4*)(p + 8);
  return x.v;
}

// ASYNCcnt-tracked 16B global->LDS copy (one lane = one b128)
DEVI void async_copy_b128(const void* gsrc, void* ldst) {
  __builtin_amdgcn_global_load_async_to_lds_b128(
      (__attribute__((address_space(1))) v4i*)(v4i*)gsrc,
      (__attribute__((address_space(3))) v4i*)(v4i*)ldst, 0, 0);
}

// ---------------------------------------------------------------------------
// K1: Gram matrices G[p][b] = Xdec (256 x HW) * Xfeat_p^T (ec x HW)^T, bf16 WMMA,
//     split-K over 4 pixel chunks with fp32 atomics. Also column sums for bias.
// ---------------------------------------------------------------------------
__global__ __launch_bounds__(128) void k_gram(
    const float* __restrict__ dec, const float* __restrict__ f0,
    const float* __restrict__ f1,  const float* __restrict__ f2,
    float* __restrict__ G, float* __restrict__ sdec, float* __restrict__ sfeat)
{
  int b  = blockIdx.y;
  int nc = blockIdx.x & 3;
  int p, mi, ni; decode224(blockIdx.x >> 2, p, mi, ni);
  int ec = ECP(p);
  const float* feat = (p == 0 ? f0 : (p == 1 ? f1 : f2)) + (size_t)b * ec * HW;
  const float* db   = dec + (size_t)b * CD * HW;

  int w = threadIdx.x >> 5, lane = threadIdx.x & 31;
  int wm = w >> 1, we = w & 1;
  int lr = lane & 15, lh = lane >> 4;
  int m0 = mi * 32 + wm * 16, e0 = ni * 32 + we * 16;

  const float* ap = db   + (size_t)(m0 + lr) * HW;
  const float* bp = feat + (size_t)(e0 + lr) * HW;

  v8f c = {};
  float sa = 0.f, sb = 0.f;
  bool doA = (p == 0 && ni == 0 && we == 0);
  bool doB = (mi == 0 && wm == 0);

  int nbase = nc * 2304;
  for (int n = nbase; n < nbase + 2304; n += 32) {
    float af[16], bf[16];
    int off = n + lh * 16;
#pragma unroll
    for (int i = 0; i < 16; i += 4) *(float4*)&af[i] = *(const float4*)(ap + off + i);
#pragma unroll
    for (int i = 0; i < 16; i += 4) *(float4*)&bf[i] = *(const float4*)(bp + off + i);
    if (doA) {
#pragma unroll
      for (int i = 0; i < 16; ++i) sa += af[i];
    }
    if (doB) {
#pragma unroll
      for (int i = 0; i < 16; ++i) sb += bf[i];
    }
    v16bf a  = cvt16(af);
    v16bf bb = cvt16(bf);
    c = __builtin_amdgcn_wmma_f32_16x16x32_bf16(false, a, false, bb, (short)0, c, false, false);
  }

  if (doA) {
    sa += __shfl_xor(sa, 16);
    if (lh == 0) atomicAdd(&sdec[b * CD + m0 + lr], sa);
  }
  if (doB) {
    sb += __shfl_xor(sb, 16);
    if (lh == 0) atomicAdd(&sfeat[SFO(p, b) + e0 + lr], sb);
  }
  float* g = G + GOFF(p, b);
#pragma unroll
  for (int r = 0; r < 8; ++r)
    atomicAdd(&g[(size_t)(m0 + r + lh * 8) * ec + e0 + lr], c[r]);
}

// ---------------------------------------------------------------------------
// K2: generic per-(b,p) fp32 GEMM  C[256 x ec] = A[256 x 256] * B[256 x ec]
//     using v_wmma_f32_16x16x4_f32. Used for P = q_w*G and M = fusion_w_p*Wv.
// ---------------------------------------------------------------------------
__global__ __launch_bounds__(128) void k_gemm256(
    const float* __restrict__ A0, const float* __restrict__ A1,
    const float* __restrict__ A2, int lda,
    const float* __restrict__ Bb, float* __restrict__ Cb)
{
  int b = blockIdx.y;
  int p, mi, ni; decode224(blockIdx.x, p, mi, ni);
  int ec = ECP(p);
  const float* A  = (p == 0 ? A0 : (p == 1 ? A1 : A2));
  const float* Bm = Bb + GOFF(p, b);
  float*       Cm = Cb + GOFF(p, b);

  int w = threadIdx.x >> 5, lane = threadIdx.x & 31;
  int wm = w >> 1, we = w & 1, lr = lane & 15, lh = lane >> 4;
  int m0 = mi * 32 + wm * 16, n0 = ni * 32 + we * 16;

  v8f c = {};
  for (int k = 0; k < 256; k += 4) {
    int ka = k + lh * 2;
    v2f a = *(const v2f*)(A + (m0 + lr) * lda + ka);  // contiguous pair -> b64
    v2f bv;
    bv.x = Bm[(size_t)ka * ec + n0 + lr];
    bv.y = Bm[(size_t)(ka + 1) * ec + n0 + lr];
    c = __builtin_amdgcn_wmma_f32_16x16x4_f32(false, a, false, bv, (short)0, c, false, false);
  }
#pragma unroll
  for (int r = 0; r < 8; ++r)
    Cm[(size_t)(m0 + r + lh * 8) * ec + n0 + lr] = c[r];
}

// ---------------------------------------------------------------------------
// K3: per-(b,p,h) logits = P_h * k_wh^T (+bias terms), softmax, attn + beff.
// ---------------------------------------------------------------------------
__global__ __launch_bounds__(128) void k_attn(
    const float* __restrict__ P,
    const float* qw0, const float* qw1, const float* qw2,
    const float* kw0, const float* kw1, const float* kw2,
    const float* qb0, const float* qb1, const float* qb2,
    const float* kb0, const float* kb1, const float* kb2,
    const float* vb0, const float* vb1, const float* vb2,
    const float* __restrict__ sdec, const float* __restrict__ sfeat,
    float* __restrict__ attn, float* __restrict__ beff)
{
  __shared__ float lg[32][33];
  __shared__ float SQ[32], SK[32];

  int id = blockIdx.x;
  int h = id & 7, p = (id >> 3) % 3, b = id / 24;
  int ec = ECP(p);
  const float* qw = (p == 0 ? qw0 : (p == 1 ? qw1 : qw2));
  const float* kw = (p == 0 ? kw0 : (p == 1 ? kw1 : kw2));
  const float* qb = (p == 0 ? qb0 : (p == 1 ? qb1 : qb2));
  const float* kb = (p == 0 ? kb0 : (p == 1 ? kb1 : kb2));
  const float* vb = (p == 0 ? vb0 : (p == 1 ? vb1 : vb2));
  const float* Pm = P + GOFF(p, b) + (size_t)h * 32 * ec;

  int tid = threadIdx.x;
  int w = tid >> 5, lane = tid & 31;
  int wm = w >> 1, we = w & 1, lr = lane & 15, lh = lane >> 4;
  int m0 = wm * 16, e0 = we * 16;

  v8f c = {};
  for (int k = 0; k < ec; k += 4) {
    int ka = k + lh * 2;
    v2f a = *(const v2f*)(Pm + (m0 + lr) * ec + ka);
    v2f bv;
    bv.x = kw[(h * 32 + e0 + lr) * ec + ka];
    bv.y = kw[(h * 32 + e0 + lr) * ec + ka + 1];
    c = __builtin_amdgcn_wmma_f32_16x16x4_f32(false, a, false, bv, (short)0, c, false, false);
  }
#pragma unroll
  for (int r = 0; r < 8; ++r) lg[m0 + r + lh * 8][e0 + lr] = c[r];
  __syncthreads();

  if (tid < 32) {
    float s = 0.f;
    const float* qr = qw + (h * 32 + tid) * CD;
    const float* sd = sdec + b * CD;
    for (int k = 0; k < CD; ++k) s += qr[k] * sd[k];
    SQ[tid] = s;
  } else if (tid < 64) {
    int e = tid - 32;
    float s = 0.f;
    const float* kr = kw + (h * 32 + e) * ec;
    const float* sf = sfeat + SFO(p, b);
    for (int k = 0; k < ec; ++k) s += kr[k] * sf[k];
    SK[e] = s;
  }
  __syncthreads();

  if (tid < 32) {
    int d = tid;
    float qbd = qb[h * 32 + d];
    float sqd = SQ[d];
    float row[32];
    float mx = -3.0e38f;
#pragma unroll
    for (int e = 0; e < 32; ++e) {
      float kbe = kb[h * 32 + e];
      float v = (lg[d][e] + qbd * SK[e] + kbe * sqd + (float)HW * qbd * kbe) * INV_SCALE;
      row[e] = v;
      mx = fmaxf(mx, v);
    }
    float sum = 0.f;
#pragma unroll
    for (int e = 0; e < 32; ++e) { row[e] = __expf(row[e] - mx); sum += row[e]; }
    float inv = 1.f / sum;
    float* ao = attn + ((size_t)((b * 3 + p) * 8 + h)) * 1024 + d * 32;
    float bs = 0.f;
#pragma unroll
    for (int e = 0; e < 32; ++e) {
      float a = row[e] * inv;
      ao[e] = a;
      bs += a * vb[h * 32 + e];
    }
    beff[(b * 3 + p) * 256 + h * 32 + d] = bs;
  }
}

// ---------------------------------------------------------------------------
// K4: Wv[p][b] rows h*32.. = attn_h (32x32) * v_w rows h*32.. (32 x ec)
// ---------------------------------------------------------------------------
__global__ __launch_bounds__(128) void k_wv(
    const float* __restrict__ attn,
    const float* vw0, const float* vw1, const float* vw2,
    float* __restrict__ Wv)
{
  int b = blockIdx.y;
  int p, h, ni; decode224(blockIdx.x, p, h, ni);
  int ec = ECP(p);
  const float* vw = (p == 0 ? vw0 : (p == 1 ? vw1 : vw2));
  const float* At = attn + ((size_t)((b * 3 + p) * 8 + h)) * 1024;

  int w = threadIdx.x >> 5, lane = threadIdx.x & 31;
  int wm = w >> 1, we = w & 1, lr = lane & 15, lh = lane >> 4;
  int m0 = wm * 16, n0 = ni * 32 + we * 16;

  v8f c = {};
  for (int k = 0; k < 32; k += 4) {
    int ka = k + lh * 2;
    v2f a = *(const v2f*)(At + (m0 + lr) * 32 + ka);
    v2f bv;
    bv.x = vw[(h * 32 + ka) * ec + n0 + lr];
    bv.y = vw[(h * 32 + ka + 1) * ec + n0 + lr];
    c = __builtin_amdgcn_wmma_f32_16x16x4_f32(false, a, false, bv, (short)0, c, false, false);
  }
  float* Wm = Wv + GOFF(p, b);
#pragma unroll
  for (int r = 0; r < 8; ++r)
    Wm[(size_t)(h * 32 + m0 + r + lh * 8) * ec + n0 + lr] = c[r];
}

// K5: total output bias per (b, out-channel)
__global__ __launch_bounds__(256) void k_btot(
    const float* __restrict__ fw, const float* __restrict__ fb,
    const float* __restrict__ beff, float* __restrict__ btot)
{
  int b = blockIdx.x, o = threadIdx.x;
  float s = fb[o];
  const float* fr = fw + o * 768;
  for (int p = 0; p < 3; ++p) {
    const float* be = beff + (b * 3 + p) * 256;
    for (int k = 0; k < 256; ++k) s += fr[p * 256 + k] * be[k];
  }
  btot[b * 256 + o] = s;
}

__global__ void k_tobf(const float* __restrict__ src, __bf16* __restrict__ dst, int n) {
  int i = blockIdx.x * blockDim.x + threadIdx.x;
  if (i < n) dst[i] = (__bf16)src[i];
}

// ---------------------------------------------------------------------------
// K6: fused[b] = sum_p M_p[b] (256 x ec) * Xfeat_p[b] (ec x HW) + btot, bf16 WMMA.
//     A tile staged via ASYNCcnt-tracked global_load_async_to_lds_b128 (bf16,
//     copy-through layout) overlapping the fp32->bf16 transposed B staging.
//     Also GN1 partial stats.
// ---------------------------------------------------------------------------
__global__ __launch_bounds__(256) void k_fused(
    const __bf16* __restrict__ Mbf, const float* __restrict__ f0,
    const float* __restrict__ f1,   const float* __restrict__ f2,
    const float* __restrict__ btot, float* __restrict__ fused,
    float* __restrict__ g1)
{
  __shared__ __bf16 sA[64][40];
  __shared__ __bf16 sB[128][40];
  __shared__ float  sbias[64];

  int b = blockIdx.z, mt = blockIdx.y, nt = blockIdx.x;
  int tid = threadIdx.x, w = tid >> 5, lane = tid & 31;
  int lr = lane & 15, lh = lane >> 4;
  int wm = w >> 2, wn = w & 3;

  if (tid < 64) sbias[tid] = btot[b * CD + mt * 64 + tid];

  v8f c[2][2] = {};
  for (int kg = 0; kg < 896; kg += 32) {
    int p  = kg < 128 ? 0 : (kg < 384 ? 1 : 2);
    int kl = kg - PREF(p);
    int ec = ECP(p);
    const __bf16* Am  = Mbf + GOFF(p, b);
    const float* feat = (p == 0 ? f0 : (p == 1 ? f1 : f2)) + (size_t)b * ec * HW;

    __syncthreads();
    {  // async-stage A: 64 rows x 32 k (bf16), one b128 per thread
      int row = tid >> 2, seg = tid & 3;
      const __bf16* gsrc = Am + (size_t)(mt * 64 + row) * ec + kl + seg * 8;
      async_copy_b128(gsrc, &sA[row][seg * 8]);
    }
    {  // stage B transposed: sB[n][k]  (fp32 load -> bf16), overlaps async A
      int kr = tid >> 3, cb = (tid & 7) * 16;
      const float* sp = feat + (size_t)(kl + kr) * HW + nt * 128 + cb;
      __builtin_prefetch(sp + 32 * (size_t)HW, 0, 1);  // next K-step rows
#pragma unroll
      for (int j = 0; j < 16; j += 4) {
        float4 v = *(const float4*)(sp + j);
        sB[cb + j + 0][kr] = (__bf16)v.x;
        sB[cb + j + 1][kr] = (__bf16)v.y;
        sB[cb + j + 2][kr] = (__bf16)v.z;
        sB[cb + j + 3][kr] = (__bf16)v.w;
      }
    }
    __builtin_amdgcn_s_wait_asynccnt(0);
    __syncthreads();

    v16bf a[2], bb[2];
#pragma unroll
    for (int i = 0; i < 2; ++i) a[i]  = ld_bf16x16(&sA[wm * 32 + i * 16 + lr][lh * 16]);
#pragma unroll
    for (int j = 0; j < 2; ++j) bb[j] = ld_bf16x16(&sB[wn * 32 + j * 16 + lr][lh * 16]);
#pragma unroll
    for (int i = 0; i < 2; ++i)
#pragma unroll
      for (int j = 0; j < 2; ++j)
        c[i][j] = __builtin_amdgcn_wmma_f32_16x16x32_bf16(
            false, a[i], false, bb[j], (short)0, c[i][j], false, false);
  }

  float s = 0.f, sq = 0.f;
#pragma unroll
  for (int i = 0; i < 2; ++i)
#pragma unroll
    for (int j = 0; j < 2; ++j)
#pragma unroll
      for (int r = 0; r < 8; ++r) {
        int rl  = wm * 32 + i * 16 + r + lh * 8;
        int row = mt * 64 + rl;
        int col = nt * 128 + wn * 32 + j * 16 + lr;
        float v = c[i][j][r] + sbias[rl];
        fused[((size_t)b * CD + row) * HW + col] = v;
        s += v; sq += v * v;
      }
#pragma unroll
  for (int off = 16; off; off >>= 1) {
    s  += __shfl_down(s, off);
    sq += __shfl_down(sq, off);
  }
  if (lane == 0) {  // all 32 rows of this wave lie in one channel-group
    int g = (mt * 64 + wm * 32) >> 5;
    atomicAdd(&g1[(b * 8 + g) * 2 + 0], s);
    atomicAdd(&g1[(b * 8 + g) * 2 + 1], sq);
  }
}

__global__ void k_statfin(float* g) {
  int i = threadIdx.x;
  if (i < 128) {
    const float invN = 1.f / (32.f * 9216.f);
    float m = g[2 * i] * invN;
    float var = g[2 * i + 1] * invN - m * m;
    g[2 * i] = m;
    g[2 * i + 1] = rsqrtf(var + EPS);
  }
}

// K7: y = dec + SiLU(GN1(fused)); writes y to d_out; GN2 partial stats
__global__ __launch_bounds__(256) void k_y(
    const float* __restrict__ fused, const float* __restrict__ dec,
    const float* __restrict__ g1, const float* __restrict__ w1,
    const float* __restrict__ b1, float* __restrict__ y, float* __restrict__ g2)
{
  size_t idx = ((size_t)blockIdx.x * 256 + threadIdx.x) * 4;
  int rowi = (int)(idx / HW);
  int b = rowi >> 8, ch = rowi & 255, g = ch >> 5;
  float m = g1[(b * 8 + g) * 2], r = g1[(b * 8 + g) * 2 + 1];
  float wc = w1[ch], bc = b1[ch];
  float4 f = *(const float4*)(fused + idx);
  float4 d = *(const float4*)(dec + idx);
  float s = 0.f, sq = 0.f;
  auto fn = [&](float fv, float dv) {
    float h = (fv - m) * r * wc + bc;
    float sl = h / (1.f + __expf(-h));
    float v = dv + sl;
    s += v; sq += v * v;
    return v;
  };
  float4 o;
  o.x = fn(f.x, d.x); o.y = fn(f.y, d.y); o.z = fn(f.z, d.z); o.w = fn(f.w, d.w);
  *(float4*)(y + idx) = o;
  int lane = threadIdx.x & 31;
#pragma unroll
  for (int off = 16; off; off >>= 1) {
    s  += __shfl_down(s, off);
    sq += __shfl_down(sq, off);
  }
  if (lane == 0) {  // 128 consecutive elems per wave stay within one (b,ch) row
    atomicAdd(&g2[(b * 8 + g) * 2 + 0], s);
    atomicAdd(&g2[(b * 8 + g) * 2 + 1], sq);
  }
}

// K8: final GN2 normalize, in-place on d_out
__global__ __launch_bounds__(256) void k_final(
    float* __restrict__ y, const float* __restrict__ g2,
    const float* __restrict__ w2, const float* __restrict__ b2)
{
  size_t idx = ((size_t)blockIdx.x * 256 + threadIdx.x) * 4;
  int rowi = (int)(idx / HW);
  int b = rowi >> 8, ch = rowi & 255, g = ch >> 5;
  float m = g2[(b * 8 + g) * 2], r = g2[(b * 8 + g) * 2 + 1];
  float wc = w2[ch] * r, bc = b2[ch];
  float4 v = *(const float4*)(y + idx);
  v.x = (v.x - m) * wc + bc;
  v.y = (v.y - m) * wc + bc;
  v.z = (v.z - m) * wc + bc;
  v.w = (v.w - m) * wc + bc;
  *(float4*)(y + idx) = v;
}

// ---------------------------------------------------------------------------
extern "C" void kernel_launch(void* const* d_in, const int* in_sizes, int n_in,
                              void* d_out, int out_size, void* d_ws, size_t ws_size,
                              hipStream_t stream) {
  (void)in_sizes; (void)n_in; (void)out_size; (void)ws_size;

  const float* dec = (const float*)d_in[0];
  const float* f0  = (const float*)d_in[1];
  const float* f1  = (const float*)d_in[2];
  const float* f2  = (const float*)d_in[3];
  const float* qw[3] = {(const float*)d_in[4],  (const float*)d_in[10], (const float*)d_in[16]};
  const float* qb[3] = {(const float*)d_in[5],  (const float*)d_in[11], (const float*)d_in[17]};
  const float* kw[3] = {(const float*)d_in[6],  (const float*)d_in[12], (const float*)d_in[18]};
  const float* kb[3] = {(const float*)d_in[7],  (const float*)d_in[13], (const float*)d_in[19]};
  const float* vw[3] = {(const float*)d_in[8],  (const float*)d_in[14], (const float*)d_in[20]};
  const float* vb[3] = {(const float*)d_in[9],  (const float*)d_in[15], (const float*)d_in[21]};
  const float* fw  = (const float*)d_in[22];
  const float* fb  = (const float*)d_in[23];
  const float* gw1 = (const float*)d_in[24];
  const float* gb1 = (const float*)d_in[25];
  const float* gw2 = (const float*)d_in[26];
  const float* gb2 = (const float*)d_in[27];

  float* ws = (float*)d_ws;
  const size_t GSZ = (size_t)16 * 256 * 896;  // 3,670,016
  float* G     = ws;
  float* P     = G + GSZ;
  float* Wv    = P + GSZ;
  float* M     = Wv + GSZ;
  float* attn  = M + GSZ;              // 393216
  float* beff  = attn + 393216;        // 12288
  float* btot  = beff + 12288;         // 4096
  float* sdec  = btot + 4096;          // 4096
  float* sfeat = sdec + 4096;          // 14336
  float* g1    = sfeat + 14336;        // 256
  float* g2    = g1 + 256;             // 256
  float* fused = g2 + 256;             // 16*256*9216
  __bf16* Mbf  = (__bf16*)(fused + (size_t)16 * 256 * 9216);

  (void)hipMemsetAsync(G, 0, GSZ * sizeof(float), stream);
  (void)hipMemsetAsync(sdec, 0, (4096 + 14336 + 256 + 256) * sizeof(float), stream);

  k_gram<<<dim3(896, 16), 128, 0, stream>>>(dec, f0, f1, f2, G, sdec, sfeat);
  k_gemm256<<<dim3(224, 16), 128, 0, stream>>>(qw[0], qw[1], qw[2], 256, G, P);
  k_attn<<<384, 128, 0, stream>>>(P, qw[0], qw[1], qw[2], kw[0], kw[1], kw[2],
                                  qb[0], qb[1], qb[2], kb[0], kb[1], kb[2],
                                  vb[0], vb[1], vb[2], sdec, sfeat, attn, beff);
  k_wv<<<dim3(224, 16), 128, 0, stream>>>(attn, vw[0], vw[1], vw[2], Wv);
  k_gemm256<<<dim3(224, 16), 128, 0, stream>>>(fw, fw + 256, fw + 512, 768, Wv, M);
  k_btot<<<16, 256, 0, stream>>>(fw, fb, beff, btot);
  k_tobf<<<(int)(GSZ / 256), 256, 0, stream>>>(M, Mbf, (int)GSZ);
  k_fused<<<dim3(72, 4, 16), 256, 0, stream>>>(Mbf, f0, f1, f2, btot, fused, g1);
  k_statfin<<<1, 128, 0, stream>>>(g1);
  float* y = (float*)d_out;
  k_y<<<36864, 256, 0, stream>>>(fused, dec, g1, gw1, gb1, y, g2);
  k_statfin<<<1, 128, 0, stream>>>(g2);
  k_final<<<36864, 256, 0, stream>>>(y, g2, gw2, gb2);
}
